// MixtureOfExperts_25099788878446
// MI455X (gfx1250) — matrix operand order repaired and checked by reading, not compiled, not run on hardware
//
#include <hip/hip_runtime.h>
#include <hip/hip_bf16.h>
#include <math.h>

// ---------------------------------------------------------------------------
// MoE forward for MI455X (gfx1250): bf16 WMMA GEMMs, top-2 routed experts,
// async global->LDS tile staging (ASYNCcnt), 64x64 WMMA tiles per wave.
// ---------------------------------------------------------------------------

typedef __attribute__((ext_vector_type(16))) __bf16 v16bf;
typedef __attribute__((ext_vector_type(8)))  float  v8f;
typedef __attribute__((ext_vector_type(4)))  int    v4i;

#define NT   4096      // B*S tokens
#define D_   1024      // model dim
#define H_   4096      // hidden dim
#define NE   8         // experts
#define PAIR_CAP 9216  // 2*NT pairs + 8*128 alignment padding
#define BM 128
#define BN 256
#define BK 32

#if defined(__AMDGCN__) && __has_builtin(__builtin_amdgcn_global_load_async_to_lds_b128)
#define HAVE_ASYNC 1
#define GLB_AS __attribute__((address_space(1)))
#define LDS_AS __attribute__((address_space(3)))
#if __has_builtin(__builtin_amdgcn_s_wait_asynccnt)
#define ASYNC_WAIT() __builtin_amdgcn_s_wait_asynccnt(0)
#else
#define ASYNC_WAIT() asm volatile("s_wait_asynccnt 0x0" ::: "memory")
#endif
#else
#define HAVE_ASYNC 0
#endif

union FragBF { uint4 q[2]; v16bf v; };

__device__ __forceinline__ unsigned short f2bf(float f) {
  // round-to-nearest-even fp32 -> bf16 (bits)
  unsigned int u = __float_as_uint(f);
  unsigned int r = (u + 0x7fffu + ((u >> 16) & 1u)) >> 16;
  return (unsigned short)r;
}

__device__ __forceinline__ float gelu_exact(float v) {
  return 0.5f * v * (1.0f + erff(v * 0.70710678118654752f));
}

// --------------------------- small helper kernels ---------------------------

__global__ void k_init(int* counts, int* cursor) {
  int t = threadIdx.x;
  if (t < NE) { counts[t] = 0; cursor[t] = 0; }
}

__global__ void k_cvt(const float* __restrict__ src, unsigned short* __restrict__ dst, int n) {
  for (int i = blockIdx.x * blockDim.x + threadIdx.x; i < n; i += gridDim.x * blockDim.x)
    dst[i] = f2bf(src[i]);
}

// Transpose-convert: src [batch][R][C] fp32 row-major -> dst [batch][C][R] bf16.
__global__ __launch_bounds__(256) void k_cvt_wt(const float* __restrict__ src,
                                                unsigned short* __restrict__ dst,
                                                int R, int C) {
  __shared__ float tile[32][33];
  const size_t base = (size_t)blockIdx.z * R * C;
  src += base; dst += base;
  const int c0 = blockIdx.x * 32, r0 = blockIdx.y * 32;
  const int tx = threadIdx.x, ty = threadIdx.y;      // block (32, 8)
#pragma unroll
  for (int i = 0; i < 4; ++i)
    tile[ty + i * 8][tx] = src[(size_t)(r0 + ty + i * 8) * C + c0 + tx];
  __syncthreads();
#pragma unroll
  for (int i = 0; i < 4; ++i)
    dst[(size_t)(c0 + ty + i * 8) * R + r0 + tx] = f2bf(tile[tx][ty + i * 8]);
}

// Router: one wave per token. logits = x@rw + rb, softmax, top-2 probs.
__global__ __launch_bounds__(256) void k_router(const float* __restrict__ x,
                                                const float* __restrict__ rw,
                                                const float* __restrict__ rb,
                                                int* __restrict__ topk_e,
                                                float* __restrict__ topk_w,
                                                int* __restrict__ counts) {
  const int wv = threadIdx.x >> 5, lane = threadIdx.x & 31;
  const int t = blockIdx.x * 8 + wv;
  float a[NE];
#pragma unroll
  for (int e = 0; e < NE; ++e) a[e] = 0.f;
  const float* xp = x + (size_t)t * D_;
  for (int i = 0; i < D_ / 32; ++i) {
    const int k = i * 32 + lane;
    const float xv = xp[k];
    const float* wp = rw + (size_t)k * NE;
#pragma unroll
    for (int e = 0; e < NE; ++e) a[e] = fmaf(xv, wp[e], a[e]);
  }
#pragma unroll
  for (int off = 16; off; off >>= 1)
#pragma unroll
    for (int e = 0; e < NE; ++e) a[e] += __shfl_xor(a[e], off, 32);
  if (lane == 0) {
    float p[NE];
    float m = -1e30f;
#pragma unroll
    for (int e = 0; e < NE; ++e) { a[e] += rb[e]; m = fmaxf(m, a[e]); }
    float s = 0.f;
#pragma unroll
    for (int e = 0; e < NE; ++e) { p[e] = __expf(a[e] - m); s += p[e]; }
    const float inv = 1.f / s;
    int i1 = 0;
#pragma unroll
    for (int e = 1; e < NE; ++e) if (p[e] > p[i1]) i1 = e;
    int i2 = (i1 == 0) ? 1 : 0;
#pragma unroll
    for (int e = 0; e < NE; ++e) if (e != i1 && p[e] > p[i2]) i2 = e;
    topk_e[t * 2 + 0] = i1; topk_w[t * 2 + 0] = p[i1] * inv;
    topk_e[t * 2 + 1] = i2; topk_w[t * 2 + 1] = p[i2] * inv;
    atomicAdd(&counts[i1], 1);
    atomicAdd(&counts[i2], 1);
  }
}

// Exclusive scan of 8 counts, offsets padded to 128-row tiles.
__global__ void k_scan(const int* __restrict__ counts, int* __restrict__ aoff) {
  if (threadIdx.x == 0) {
    int o = 0;
#pragma unroll
    for (int e = 0; e < NE; ++e) { aoff[e] = o; o += (counts[e] + BM - 1) & ~(BM - 1); }
    aoff[NE] = o;
  }
}

__global__ void k_scatter(const int* __restrict__ topk_e, const float* __restrict__ topk_w,
                          const int* __restrict__ aoff, int* __restrict__ cursor,
                          int* __restrict__ pair_token, float* __restrict__ pair_w) {
  const int t = blockIdx.x * blockDim.x + threadIdx.x;
  if (t >= NT) return;
#pragma unroll
  for (int k = 0; k < 2; ++k) {
    const int e = topk_e[t * 2 + k];
    const int pos = atomicAdd(&cursor[e], 1);
    const int slot = aoff[e] + pos;
    pair_token[slot] = t;
    pair_w[slot] = topk_w[t * 2 + k];
  }
}

// ------------------------------ WMMA GEMM ----------------------------------
// C[M,N] = A[M,K] * Wt[N,K]^T, bf16 inputs, f32 accumulate.
// Block tile 128x256, 8 waves as 2(m) x 4(n), wave tile 64x64 = 4x4 WMMA tiles.
// MODE 0: shared FFN1  : A = xbf rows direct,  epilog GELU -> bf16 Hs
// MODE 1: expert FFN1  : A = xbf gathered via pair_token, GELU -> bf16 He
// MODE 2: shared FFN2  : A = Hs direct, epilog  out = v + bias (plain store)
// MODE 3: expert FFN2  : A = He direct, epilog  atomicAdd(out[token], w*(v+bias))
template <int MODE>
__global__ __launch_bounds__(256, 2) void moe_gemm(const unsigned short* __restrict__ A,
                                                   const unsigned short* __restrict__ Wt,
                                                   const float* __restrict__ bias,
                                                   float* __restrict__ outF,
                                                   unsigned short* __restrict__ outH,
                                                   const int* __restrict__ pair_token,
                                                   const float* __restrict__ pair_w,
                                                   const int* __restrict__ aoff,
                                                   const int* __restrict__ counts,
                                                   int K, int N) {
  __shared__ unsigned short As[2][BM * BK];   // 2 x 8 KB
  __shared__ unsigned short Bs[2][BN * BK];   // 2 x 16 KB

  const int row0 = blockIdx.x * BM;
  const int n0   = blockIdx.y * BN;

  const unsigned short* Wp = Wt;
  const float* bp = bias;
  int row_end = 1 << 30;

  if (MODE == 1 || MODE == 3) {
    int e = -1;
#pragma unroll
    for (int i = 0; i < NE; ++i)
      if (row0 >= aoff[i] && row0 < aoff[i + 1]) e = i;
    if (e < 0) return;                       // beyond last expert's region
    row_end = aoff[e] + counts[e];
    if (row0 >= row_end) return;             // 128-alignment padding hole
    Wp = Wt + (size_t)e * N * K;
    bp = bias + (size_t)e * N;
  }

  const int tid  = threadIdx.x;
  const int lane = tid & 31;
  const int wv   = tid >> 5;
  const int wm   = (wv & 1) * 64;            // wave row offset in tile (2 m-waves)
  const int wn   = (wv >> 1) * 64;           // wave col offset in tile (4 n-waves)
  const int lm   = lane & 15;
  const int lh   = lane >> 4;

  // Loop-invariant source rows for the tile fill (A: 512 16B chunks, B: 1024).
  int arow[2], brow[4];
#pragma unroll
  for (int i = 0; i < 2; ++i) {
    const int r = (tid + i * 256) >> 2;
    if (MODE == 1) {
      const int gr = row0 + r;
      arow[i] = (gr < row_end) ? pair_token[gr] : 0;     // gather token rows of x
    } else {
      arow[i] = row0 + r;
    }
  }
#pragma unroll
  for (int i = 0; i < 4; ++i) brow[i] = n0 + ((tid + i * 256) >> 2);

#if HAVE_ASYNC
  // gfx1250 path: DMA the tiles straight into LDS, tracked with ASYNCcnt.
  auto issue = [&](int kt, int buf) {
    const int k0 = kt * BK;
#pragma unroll
    for (int i = 0; i < 2; ++i) {
      const int c = tid + i * 256;
      const int kc = (c & 3) * 8;
      __builtin_amdgcn_global_load_async_to_lds_b128(
          (GLB_AS v4i*)(A + (size_t)arow[i] * K + k0 + kc),
          (LDS_AS v4i*)&As[buf][8 * c], 0, 0);
    }
#pragma unroll
    for (int i = 0; i < 4; ++i) {
      const int c = tid + i * 256;
      const int kc = (c & 3) * 8;
      __builtin_amdgcn_global_load_async_to_lds_b128(
          (GLB_AS v4i*)(Wp + (size_t)brow[i] * K + k0 + kc),
          (LDS_AS v4i*)&Bs[buf][8 * c], 0, 0);
    }
  };
#else
  uint4 areg[2], breg[4];
  auto loadG = [&](int kt) {
    const int k0 = kt * BK;
#pragma unroll
    for (int i = 0; i < 2; ++i) {
      const int c = tid + i * 256;
      const int kc = (c & 3) * 8;
      areg[i] = *(const uint4*)(A + (size_t)arow[i] * K + k0 + kc);
    }
#pragma unroll
    for (int i = 0; i < 4; ++i) {
      const int c = tid + i * 256;
      const int kc = (c & 3) * 8;
      breg[i] = *(const uint4*)(Wp + (size_t)brow[i] * K + k0 + kc);
    }
  };
  auto storeL = [&](int buf) {
#pragma unroll
    for (int i = 0; i < 2; ++i) *(uint4*)(&As[buf][8 * (tid + i * 256)]) = areg[i];
#pragma unroll
    for (int i = 0; i < 4; ++i) *(uint4*)(&Bs[buf][8 * (tid + i * 256)]) = breg[i];
  };
#endif

  const v8f vzero = {0.f, 0.f, 0.f, 0.f, 0.f, 0.f, 0.f, 0.f};
  v8f acc[4][4];
#pragma unroll
  for (int mt = 0; mt < 4; ++mt)
#pragma unroll
    for (int nt = 0; nt < 4; ++nt) acc[mt][nt] = vzero;

  const int KT = K / BK;

#if HAVE_ASYNC
  issue(0, 0);
  ASYNC_WAIT();
  __syncthreads();
#else
  loadG(0);
  storeL(0);
  __syncthreads();
#endif

  for (int kt = 0; kt < KT; ++kt) {
    const int buf = kt & 1;
#if HAVE_ASYNC
    if (kt + 1 < KT) issue(kt + 1, buf ^ 1);
#else
    if (kt + 1 < KT) loadG(kt + 1);
#endif

    FragBF afr[4], bfr[4];
#pragma unroll
    for (int mt = 0; mt < 4; ++mt) {
      // A 16x32 bf16 fragment: lane holds row lm, K in [8*lh,+8) and [16+8*lh,+8)
      const unsigned short* p = &As[buf][(wm + mt * 16 + lm) * BK];
      afr[mt].q[0] = *(const uint4*)(p + 8 * lh);
      afr[mt].q[1] = *(const uint4*)(p + 16 + 8 * lh);
    }
#pragma unroll
    for (int nt = 0; nt < 4; ++nt) {
      // B 32x16 bf16 fragment: lane holds col lm, K in [8*lh,+8) and [16+8*lh,+8)
      const unsigned short* p = &Bs[buf][(wn + nt * 16 + lm) * BK];
      bfr[nt].q[0] = *(const uint4*)(p + 8 * lh);
      bfr[nt].q[1] = *(const uint4*)(p + 16 + 8 * lh);
    }
#pragma unroll
    for (int mt = 0; mt < 4; ++mt)
#pragma unroll
      for (int nt = 0; nt < 4; ++nt)
        acc[mt][nt] = __builtin_amdgcn_wmma_f32_16x16x32_bf16(
            false, afr[mt].v, false, bfr[nt].v, (short)0, acc[mt][nt], false, false);

#if HAVE_ASYNC
    if (kt + 1 < KT) ASYNC_WAIT();
#else
    if (kt + 1 < KT) storeL(buf ^ 1);
#endif
    __syncthreads();
  }

  // Epilogue. C layout: vgpr j -> m = 8*lh + j, n = lm (within each 16x16 tile).
  float bv[4];
#pragma unroll
  for (int nt = 0; nt < 4; ++nt) bv[nt] = bp[n0 + wn + nt * 16 + lm];

#pragma unroll
  for (int mt = 0; mt < 4; ++mt) {
#pragma unroll
    for (int j = 0; j < 8; ++j) {
      const int r = row0 + wm + mt * 16 + lh * 8 + j;
      if ((MODE == 1 || MODE == 3) && r >= row_end) continue;
      int tok = 0;
      float wgt = 0.f;
      if (MODE == 3) { tok = pair_token[r]; wgt = pair_w[r]; }
#pragma unroll
      for (int nt = 0; nt < 4; ++nt) {
        const int cc = n0 + wn + nt * 16 + lm;
        const float v = acc[mt][nt][j] + bv[nt];
        if (MODE == 0 || MODE == 1) {
          outH[(size_t)r * N + cc] = f2bf(gelu_exact(v));
        } else if (MODE == 2) {
          outF[(size_t)r * N + cc] = v;
        } else {
          atomicAdd(&outF[(size_t)tok * N + cc], wgt * v);
        }
      }
    }
  }
  (void)counts;
}

// ------------------------------- launcher ----------------------------------

extern "C" void kernel_launch(void* const* d_in, const int* in_sizes, int n_in,
                              void* d_out, int out_size, void* d_ws, size_t ws_size,
                              hipStream_t stream) {
  (void)in_sizes; (void)n_in; (void)out_size; (void)ws_size;
  const float* x        = (const float*)d_in[0];
  const float* router_w = (const float*)d_in[1];
  const float* router_b = (const float*)d_in[2];
  const float* w1       = (const float*)d_in[3];
  const float* b1       = (const float*)d_in[4];
  const float* w2       = (const float*)d_in[5];
  const float* b2       = (const float*)d_in[6];
  const float* sw1      = (const float*)d_in[7];
  const float* sb1      = (const float*)d_in[8];
  const float* sw2      = (const float*)d_in[9];
  const float* sb2      = (const float*)d_in[10];
  float* out = (float*)d_out;

  char* ws = (char*)d_ws;
  size_t off = 0;
  auto take = [&](size_t bytes) -> void* {
    off = (off + 255) & ~(size_t)255;
    void* p = ws + off;
    off += bytes;
    return p;
  };
  unsigned short* xbf  = (unsigned short*)take((size_t)NT * D_ * 2);
  unsigned short* sw1t = (unsigned short*)take((size_t)H_ * D_ * 2);
  unsigned short* sw2t = (unsigned short*)take((size_t)D_ * H_ * 2);
  unsigned short* w1t  = (unsigned short*)take((size_t)NE * H_ * D_ * 2);
  unsigned short* w2t  = (unsigned short*)take((size_t)NE * D_ * H_ * 2);
  unsigned short* Hs   = (unsigned short*)take((size_t)NT * H_ * 2);
  unsigned short* He   = (unsigned short*)take((size_t)PAIR_CAP * H_ * 2);
  int*   topk_e     = (int*)take((size_t)NT * 2 * 4);
  float* topk_w     = (float*)take((size_t)NT * 2 * 4);
  int*   counts     = (int*)take(NE * 4);
  int*   cursor     = (int*)take(NE * 4);
  int*   aoff       = (int*)take((NE + 1) * 4);
  int*   pair_token = (int*)take((size_t)PAIR_CAP * 4);
  float* pair_w     = (float*)take((size_t)PAIR_CAP * 4);

  const dim3 tb(32, 8);

  k_init<<<1, 32, 0, stream>>>(counts, cursor);
  k_cvt<<<2048, 256, 0, stream>>>(x, xbf, NT * D_);
  k_cvt_wt<<<dim3(H_ / 32, D_ / 32, 1),  tb, 0, stream>>>(sw1, sw1t, D_, H_);
  k_cvt_wt<<<dim3(D_ / 32, H_ / 32, 1),  tb, 0, stream>>>(sw2, sw2t, H_, D_);
  k_cvt_wt<<<dim3(H_ / 32, D_ / 32, NE), tb, 0, stream>>>(w1, w1t, D_, H_);
  k_cvt_wt<<<dim3(D_ / 32, H_ / 32, NE), tb, 0, stream>>>(w2, w2t, H_, D_);
  k_router<<<NT / 8, 256, 0, stream>>>(x, router_w, router_b, topk_e, topk_w, counts);
  k_scan<<<1, 32, 0, stream>>>(counts, aoff);
  k_scatter<<<NT / 256, 256, 0, stream>>>(topk_e, topk_w, aoff, cursor, pair_token, pair_w);

  // Shared FFN: Hs = gelu(x@sw1+sb1); out = Hs@sw2 + sb2
  moe_gemm<0><<<dim3(NT / BM, H_ / BN), 256, 0, stream>>>(
      xbf, sw1t, sb1, nullptr, Hs, nullptr, nullptr, nullptr, nullptr, D_, H_);
  // Expert FFN1: He[slot] = gelu(x[token]@w1[e]+b1[e])
  moe_gemm<1><<<dim3(PAIR_CAP / BM, H_ / BN), 256, 0, stream>>>(
      xbf, w1t, b1, nullptr, He, pair_token, nullptr, aoff, counts, D_, H_);
  // Shared FFN2 writes out first (plain stores), then experts accumulate.
  moe_gemm<2><<<dim3(NT / BM, D_ / BN), 256, 0, stream>>>(
      Hs, sw2t, sb2, out, nullptr, nullptr, nullptr, nullptr, nullptr, H_, D_);
  moe_gemm<3><<<dim3(PAIR_CAP / BM, D_ / BN), 256, 0, stream>>>(
      He, w2t, b2, out, nullptr, pair_token, pair_w, aoff, counts, H_, D_);
}
